// MoEEncoderBlock_84928683311112
// MI455X (gfx1250) — matrix operand order repaired and checked by reading, not compiled, not run on hardware
//
#include <hip/hip_runtime.h>
#include <hip/hip_bf16.h>

#define D_MODEL 1280
#define NUM_EXPERTS 20
#define HEAD_DIM 64
#define QKV3 (3 * HEAD_DIM) /* 192 */
#define D_FF 5120
#define BATCH 2
#define SEQ 2048
#define BS (BATCH * SEQ) /* 4096 */

typedef __attribute__((ext_vector_type(16))) __bf16 bf16x16;
typedef __attribute__((ext_vector_type(8))) float f32x8;
typedef __attribute__((ext_vector_type(4))) unsigned int u32x4;
typedef __attribute__((ext_vector_type(8))) int i32x8;
typedef __attribute__((ext_vector_type(4))) int i32x4;

#if defined(__HIP_DEVICE_COMPILE__) && __has_builtin(__builtin_amdgcn_tensor_load_to_lds)
#define HAVE_TDM 1
#else
#define HAVE_TDM 0
#endif

#if defined(__HIP_DEVICE_COMPILE__) && __has_builtin(__builtin_amdgcn_update_dpp)
#define HAVE_DPP 1
#else
#define HAVE_DPP 0
#endif

// ---------- helpers ----------

__device__ __forceinline__ unsigned short f2b(float f) {
  unsigned int u = __float_as_uint(f);
  u += 0x7FFFu + ((u >> 16) & 1u); // round-to-nearest-even
  return (unsigned short)(u >> 16);
}

union FragU {
  bf16x16 v;
  u32x4 q[2];
};

union Vec8U {
  u32x4 v;
  unsigned short s[8];
};

__device__ __forceinline__ bf16x16 make_frag(u32x4 lo, u32x4 hi) {
  FragU f;
  f.q[0] = lo;
  f.q[1] = hi;
  return f.v;
}

// A-fragment (16x32 bf16, CDNA5 layout): lane l<16: M=l, K {0..7,16..23};
// lane l>=16: M=l-16, K {8..15,24..31}.  Two aligned 16B reads.
__device__ __forceinline__ bf16x16 load_a_frag(const unsigned short* base, int lane, int ldk) {
  const int m = lane & 15;
  const int kb = (lane < 16) ? 0 : 8;
  const unsigned short* r = base + (size_t)m * ldk;
  return make_frag(*(const u32x4*)(r + kb), *(const u32x4*)(r + 16 + kb));
}

// B-fragment (32x16 bf16, CDNA5 layout) from an N-major (transposed) tile.
__device__ __forceinline__ bf16x16 load_bT_frag(const unsigned short* baseT, int lane, int ldk,
                                                int nbase) {
  const int n = nbase + (lane & 15);
  const int kb = (lane < 16) ? 0 : 8;
  const unsigned short* r = baseT + (size_t)n * ldk + kb;
  return make_frag(*(const u32x4*)(r), *(const u32x4*)(r + 16));
}

__device__ __forceinline__ f32x8 wmma_bf16(bf16x16 a, bf16x16 b, f32x8 c) {
  return __builtin_amdgcn_wmma_f32_16x16x32_bf16(false, a, false, b, (short)0, c, false, false);
}

// ---------- DPP 16-lane butterfly reductions (masks {1,2,7,15} span GF(2)^4)

template <int CTRL>
__device__ __forceinline__ float dppf(float x) {
#if HAVE_DPP
  return __int_as_float(
      __builtin_amdgcn_update_dpp(0, __float_as_int(x), CTRL, 0xF, 0xF, true));
#else
  return x;
#endif
}

__device__ __forceinline__ float red_max16(float x) {
#if HAVE_DPP
  x = fmaxf(x, dppf<0xB1>(x));  // quad_perm(1,0,3,2)  = xor 1
  x = fmaxf(x, dppf<0x4E>(x));  // quad_perm(2,3,0,1)  = xor 2
  x = fmaxf(x, dppf<0x141>(x)); // row_half_mirror     = xor 7
  x = fmaxf(x, dppf<0x140>(x)); // row_mirror          = xor 15
#else
  x = fmaxf(x, __shfl_xor(x, 1));
  x = fmaxf(x, __shfl_xor(x, 2));
  x = fmaxf(x, __shfl_xor(x, 4));
  x = fmaxf(x, __shfl_xor(x, 8));
#endif
  return x;
}

__device__ __forceinline__ float red_sum16(float x) {
#if HAVE_DPP
  x += dppf<0xB1>(x);
  x += dppf<0x4E>(x);
  x += dppf<0x141>(x);
  x += dppf<0x140>(x);
#else
  x += __shfl_xor(x, 1);
  x += __shfl_xor(x, 2);
  x += __shfl_xor(x, 4);
  x += __shfl_xor(x, 8);
#endif
  return x;
}

// ---------- Tensor Data Mover: 2D bf16 tile -> LDS (padded rows) ----------
// D# per CDNA5 ISA §8.3-8.5.  group0: count=1 | lds_addr | global_addr(57b) | type=2.
// group1: data_size=1(2B), pad fields, tensor dims (huge; tiles always in-bounds),
// tile_dim0/1, tensor_dim0_stride.  Groups 2/3 zero (2D tensor).
#if HAVE_TDM
__device__ __forceinline__ void tdm_load_2d_bf16(unsigned int ldsOff, const void* gp,
                                                 unsigned int tileX, unsigned int tileY,
                                                 unsigned int strideElems,
                                                 unsigned int padIntervalCode, // 2^(c+1) DWORDs
                                                 unsigned int padAmountCode) { // (a+1) DWORDs
  const unsigned long long ga = (unsigned long long)gp;
  u32x4 g0;
  g0.x = 1u; // count = 1 (valid user descriptor)
  g0.y = ldsOff;
  g0.z = (unsigned int)ga;
  g0.w = (unsigned int)(ga >> 32) | (2u << 30); // type = 2 ("image")
  i32x8 g1;
  g1[0] = (int)((1u << 16) | (1u << 20) | (padIntervalCode << 22) | (padAmountCode << 25));
  g1[1] = (int)(0xFFFFu << 16);               // tensor_dim0[15:0] (huge)
  g1[2] = (int)(0x7FFFu | (0xFFFFu << 16));   // tensor_dim0 hi | tensor_dim1 lo
  g1[3] = (int)(0x7FFFu | (tileX << 16));     // tensor_dim1 hi | tile_dim0
  g1[4] = (int)(tileY & 0xFFFFu);             // tile_dim1 (tile_dim2 = 0)
  g1[5] = (int)strideElems;                   // tensor_dim0_stride[31:0]
  g1[6] = 0;
  g1[7] = 0;
  const i32x4 z4 = {0, 0, 0, 0};
#if defined(__clang_major__) && __clang_major__ >= 23
  const i32x8 z8 = {0, 0, 0, 0, 0, 0, 0, 0};
  __builtin_amdgcn_tensor_load_to_lds(g0, g1, z4, z4, z8, 0);
#else
  __builtin_amdgcn_tensor_load_to_lds(g0, g1, z4, z4, 0);
#endif
}
#endif

// ---------- elementwise cast f32 -> bf16 ----------

__global__ void cast_kernel(const float* __restrict__ src, unsigned short* __restrict__ dst,
                            long n) {
  long i = (long)blockIdx.x * blockDim.x + threadIdx.x;
  const long stride = (long)gridDim.x * blockDim.x;
  for (; i < n; i += stride) dst[i] = f2b(src[i]);
}

// ---------- router: logits + softmax, one wave per token ----------

__global__ __launch_bounds__(32) void router_kernel(const float* __restrict__ x,
                                                    const float* __restrict__ W,
                                                    const float* __restrict__ bb,
                                                    float* __restrict__ routing) {
  const int t = blockIdx.x;
  const int lane = threadIdx.x;
  const float* xr = x + (size_t)t * D_MODEL;
  float acc[NUM_EXPERTS];
#pragma unroll
  for (int e2 = 0; e2 < NUM_EXPERTS; ++e2) acc[e2] = 0.f;
  for (int d = lane; d < D_MODEL; d += 32) {
    const float xv = xr[d];
    const float* wr = W + (size_t)d * NUM_EXPERTS;
#pragma unroll
    for (int e2 = 0; e2 < NUM_EXPERTS; ++e2) acc[e2] += xv * wr[e2];
  }
#pragma unroll
  for (int e2 = 0; e2 < NUM_EXPERTS; ++e2) {
    float v = acc[e2];
    v += __shfl_xor(v, 1);
    v += __shfl_xor(v, 2);
    v += __shfl_xor(v, 4);
    v += __shfl_xor(v, 8);
    v += __shfl_xor(v, 16);
    acc[e2] = v + bb[e2];
  }
  float mx = -3.0e38f;
#pragma unroll
  for (int e2 = 0; e2 < NUM_EXPERTS; ++e2) mx = fmaxf(mx, acc[e2]);
  float sum = 0.f;
#pragma unroll
  for (int e2 = 0; e2 < NUM_EXPERTS; ++e2) {
    acc[e2] = __expf(acc[e2] - mx);
    sum += acc[e2];
  }
  if (lane == 0) {
    const float inv = 1.f / sum;
#pragma unroll
    for (int e2 = 0; e2 < NUM_EXPERTS; ++e2)
      routing[(size_t)t * NUM_EXPERTS + e2] = acc[e2] * inv;
  }
}

// ---------- generic bf16 WMMA GEMM, 128x64 tile, 8 waves ----------
// MODE 0: plain C[M,N] epilogue.  MODE 1: QKV split per expert (blockIdx.z).
// A tile staged via TDM (wave 0 issues, s_wait_tensorcnt, barrier publishes).

template <int MODE, int ACT, int RES, int WF, int WB>
__global__ __launch_bounds__(256) void gemm_kernel(
    const unsigned short* __restrict__ A, const unsigned short* __restrict__ Bg,
    const float* __restrict__ bias, const float* __restrict__ res, float* __restrict__ outF,
    unsigned short* __restrict__ outB, unsigned short* __restrict__ qp,
    unsigned short* __restrict__ kp, unsigned short* __restrict__ vp, int M, int N, int K) {
  __shared__ unsigned short As[128][40]; // 128x32 tile, row = 32 data + 8 pad ushorts
  __shared__ unsigned short Bt[64][40];  // transposed 32x64 tile -> [n][k]

  const int tid = threadIdx.x;
  const int wave = tid >> 5;
  const int lane = tid & 31;
  const int rowbase = blockIdx.y * 128;
  const int nbase = blockIdx.x * 64;

  const unsigned short* Bptr = Bg;
  if constexpr (MODE == 1) Bptr = Bg + (size_t)blockIdx.z * K * N;

  f32x8 acc[4] = {};

  for (int kt = 0; kt < K; kt += 32) {
    __syncthreads();
#if HAVE_TDM
    if (wave == 0) { // TDM: A tile (rows of 16 DWORDs, pad 4 DWORDs -> codes 3,3)
      tdm_load_2d_bf16((unsigned int)(size_t)&As[0][0], A + (size_t)rowbase * K + kt, 32u,
                       128u, (unsigned int)K, 3u, 3u);
    }
#else
#pragma unroll
    for (int i = 0; i < 2; ++i) {
      const int cidx = tid + i * 256;
      const int r = cidx >> 2, ch = cidx & 3;
      *(u32x4*)(&As[r][ch * 8]) = *(const u32x4*)(A + (size_t)(rowbase + r) * K + kt + ch * 8);
    }
#endif
    // B tile: 32 K-rows x 64 N, transposed on store
    {
      const int kr = tid >> 3, nc = tid & 7;
      Vec8U uu;
      uu.v = *(const u32x4*)(Bptr + (size_t)(kt + kr) * N + nbase + nc * 8);
#pragma unroll
      for (int j = 0; j < 8; ++j) Bt[nc * 8 + j][kr] = uu.s[j];
    }
    if (kt + 32 < K) { // hint next B K-tile into cache (global_prefetch_b8)
      __builtin_prefetch(Bptr + (size_t)(kt + 32 + (tid >> 3)) * N + nbase + (tid & 7) * 8, 0, 3);
    }
#if HAVE_TDM
    if (wave == 0) __builtin_amdgcn_s_wait_tensorcnt(0);
#endif
    __syncthreads();

    const bf16x16 af = load_a_frag(&As[wave * 16][0], lane, 40);
#pragma unroll
    for (int g = 0; g < 4; ++g) {
      const bf16x16 bf = load_bT_frag(&Bt[0][0], lane, 40, g * 16);
      acc[g] = wmma_bf16(af, bf, acc[g]);
    }
  }

  // epilogue (C layout: reg r -> M = r + 8*(lane>=16), lane&15 -> N)
  const int rowoff = (lane >> 4) << 3;
  const int ln = lane & 15;
#pragma unroll
  for (int g = 0; g < 4; ++g) {
#pragma unroll
    for (int r = 0; r < 8; ++r) {
      const int mrow = rowbase + wave * 16 + r + rowoff;
      const int ncol = nbase + g * 16 + ln;
      float v = acc[g][r];
      if constexpr (MODE == 1) {
        const int e = blockIdx.z;
        v += bias[e * QKV3 + ncol];
        const int bidx = mrow >> 11;       // / SEQ
        const int srow = mrow & (SEQ - 1); // % SEQ
        unsigned short* dst = (blockIdx.x == 0) ? qp : (blockIdx.x == 1) ? kp : vp;
        const int h = ncol & (HEAD_DIM - 1);
        dst[(((size_t)(bidx * NUM_EXPERTS + e)) * SEQ + srow) * HEAD_DIM + h] = f2b(v);
      } else {
        v += bias[ncol];
        if constexpr (RES) v += res[(size_t)mrow * N + ncol];
        if constexpr (ACT == 1) v = 0.5f * v * (1.0f + erff(v * 0.70710678118f)); // exact GELU
        if constexpr (WF) outF[(size_t)mrow * N + ncol] = v;
        if constexpr (WB) outB[(size_t)mrow * N + ncol] = f2b(v);
      }
    }
  }
}

// ---------- flash attention per (b, e, 64-row block), 4 waves ----------

__global__ __launch_bounds__(128) void attn_kernel(const unsigned short* __restrict__ qg,
                                                   const unsigned short* __restrict__ kg,
                                                   const unsigned short* __restrict__ vg,
                                                   const float* __restrict__ routing,
                                                   unsigned short* __restrict__ comb) {
  __shared__ unsigned short Ks[32][80];    // K chunk natural: [key][h] (row 64 + 16 pad)
  __shared__ unsigned short Vt[64][40];    // V chunk transposed: [h][key]
  __shared__ unsigned short Pl[4][16][40]; // per-wave P scratch, C->A layout transpose

  const int tid = threadIdx.x;
  const int wave = tid >> 5;
  const int lane = tid & 31;
  const int b = blockIdx.z, e = blockIdx.y;
  const int qbase = blockIdx.x * 64 + wave * 16;
  const size_t headbase = ((size_t)(b * NUM_EXPERTS + e)) * SEQ;

  const unsigned short* qrow = qg + (headbase + qbase) * HEAD_DIM;
  const bf16x16 qf0 = load_a_frag(qrow, lane, HEAD_DIM);      // h 0..31
  const bf16x16 qf1 = load_a_frag(qrow + 32, lane, HEAD_DIM); // h 32..63

  f32x8 o0 = {}, o1 = {}, o2 = {}, o3 = {};
  float rowm[8], rowl[8];
#pragma unroll
  for (int r = 0; r < 8; ++r) {
    rowm[r] = -3.0e38f;
    rowl[r] = 0.f;
  }
  const int rowoff = (lane >> 4) << 3;
  const int ln = lane & 15;

  for (int it = 0; it < SEQ / 32; ++it) {
    __syncthreads();
#if HAVE_TDM
    if (wave == 0) { // TDM: K tile (rows of 32 DWORDs, pad 8 DWORDs -> codes 4,7)
      tdm_load_2d_bf16((unsigned int)(size_t)&Ks[0][0],
                       kg + (headbase + it * 32) * HEAD_DIM, 64u, 32u, HEAD_DIM, 4u, 7u);
    }
#else
#pragma unroll
    for (int i = 0; i < 2; ++i) {
      const int cidx = tid + i * 128;
      const int key = cidx >> 3, hc = cidx & 7;
      *(u32x4*)(&Ks[key][hc * 8]) =
          *(const u32x4*)(kg + (headbase + it * 32 + key) * HEAD_DIM + hc * 8);
    }
#endif
    { // V chunk transposed; both loads issued before scattering
      const int cidx0 = tid, cidx1 = tid + 128;
      const int key0 = cidx0 >> 3, hc0 = cidx0 & 7;
      const int key1 = cidx1 >> 3, hc1 = cidx1 & 7;
      Vec8U uu0, uu1;
      uu0.v = *(const u32x4*)(vg + (headbase + it * 32 + key0) * HEAD_DIM + hc0 * 8);
      uu1.v = *(const u32x4*)(vg + (headbase + it * 32 + key1) * HEAD_DIM + hc1 * 8);
#pragma unroll
      for (int j = 0; j < 8; ++j) Vt[hc0 * 8 + j][key0] = uu0.s[j];
#pragma unroll
      for (int j = 0; j < 8; ++j) Vt[hc1 * 8 + j][key1] = uu1.s[j];
    }
#if HAVE_TDM
    if (wave == 0) __builtin_amdgcn_s_wait_tensorcnt(0);
#endif
    __syncthreads();

    // scores: two 16-key groups, K-dim = H = 64 -> 2 chained WMMAs each
    f32x8 s0 = {}, s1 = {};
    s0 = wmma_bf16(qf0, load_bT_frag(&Ks[0][0], lane, 80, 0), s0);
    s0 = wmma_bf16(qf1, load_bT_frag(&Ks[0][32], lane, 80, 0), s0);
    s1 = wmma_bf16(qf0, load_bT_frag(&Ks[0][0], lane, 80, 16), s1);
    s1 = wmma_bf16(qf1, load_bT_frag(&Ks[0][32], lane, 80, 16), s1);

    // online softmax; row reductions via DPP16 butterflies (no LDS traffic)
#pragma unroll
    for (int r = 0; r < 8; ++r) {
      const float a = s0[r] * 0.125f;
      const float c = s1[r] * 0.125f;
      const float mg = red_max16(fmaxf(a, c));
      const float mn = fmaxf(rowm[r], mg);
      const float al = __expf(rowm[r] - mn);
      rowm[r] = mn;
      const float p0 = __expf(a - mn);
      const float p1 = __expf(c - mn);
      rowl[r] = rowl[r] * al + p0 + p1; // lane-partial row sum
      o0[r] *= al;
      o1[r] *= al;
      o2[r] *= al;
      o3[r] *= al;
      Pl[wave][rowoff + r][ln] = f2b(p0);
      Pl[wave][rowoff + r][16 + ln] = f2b(p1);
    }

    // P (A layout via per-wave LDS scratch) @ V
    const bf16x16 pf = load_a_frag(&Pl[wave][0][0], lane, 40);
    o0 = wmma_bf16(pf, load_bT_frag(&Vt[0][0], lane, 40, 0), o0);
    o1 = wmma_bf16(pf, load_bT_frag(&Vt[0][0], lane, 40, 16), o1);
    o2 = wmma_bf16(pf, load_bT_frag(&Vt[0][0], lane, 40, 32), o2);
    o3 = wmma_bf16(pf, load_bT_frag(&Vt[0][0], lane, 40, 48), o3);
  }

#pragma unroll
  for (int r = 0; r < 8; ++r) {
    const float l = red_sum16(rowl[r]);
    const int srow = qbase + rowoff + r;
    const float wrt = routing[((size_t)(b * SEQ + srow)) * NUM_EXPERTS + e];
    const float sc = wrt / l;
    const size_t obase = ((size_t)(b * SEQ + srow)) * D_MODEL + e * HEAD_DIM;
    comb[obase + 0 + ln] = f2b(o0[r] * sc);
    comb[obase + 16 + ln] = f2b(o1[r] * sc);
    comb[obase + 32 + ln] = f2b(o2[r] * sc);
    comb[obase + 48 + ln] = f2b(o3[r] * sc);
  }
}

// ---------- layernorm per row ----------

__global__ __launch_bounds__(128) void ln_kernel(const float* __restrict__ in,
                                                 const float* __restrict__ w,
                                                 const float* __restrict__ bb,
                                                 float* __restrict__ outF,
                                                 unsigned short* __restrict__ outB) {
  __shared__ float rs[4], rs2[4];
  const int row = blockIdx.x, tid = threadIdx.x;
  const int wave = tid >> 5, lane = tid & 31;
  const float* xr = in + (size_t)row * D_MODEL;
  float s = 0.f, s2 = 0.f;
  for (int d = tid; d < D_MODEL; d += 128) {
    const float v = xr[d];
    s += v;
    s2 += v * v;
  }
#pragma unroll
  for (int off = 16; off >= 1; off >>= 1) {
    s += __shfl_xor(s, off);
    s2 += __shfl_xor(s2, off);
  }
  if (lane == 0) {
    rs[wave] = s;
    rs2[wave] = s2;
  }
  __syncthreads();
  s = rs[0] + rs[1] + rs[2] + rs[3];
  s2 = rs2[0] + rs2[1] + rs2[2] + rs2[3];
  const float mu = s * (1.f / D_MODEL);
  const float var = s2 * (1.f / D_MODEL) - mu * mu;
  const float inv = rsqrtf(var + 1e-5f);
  for (int d = tid; d < D_MODEL; d += 128) {
    const float v = (xr[d] - mu) * inv * w[d] + bb[d];
    if (outF) outF[(size_t)row * D_MODEL + d] = v;
    if (outB) outB[(size_t)row * D_MODEL + d] = f2b(v);
  }
}

// ---------- host launcher ----------

extern "C" void kernel_launch(void* const* d_in, const int* in_sizes, int n_in, void* d_out,
                              int out_size, void* d_ws, size_t ws_size, hipStream_t stream) {
  (void)in_sizes;
  (void)n_in;
  (void)out_size;
  (void)ws_size;
  const float* x = (const float*)d_in[0];
  const float* router_W = (const float*)d_in[1];
  const float* router_b = (const float*)d_in[2];
  const float* expert_W = (const float*)d_in[3];
  const float* expert_b = (const float*)d_in[4];
  const float* out_W = (const float*)d_in[5];
  const float* out_b = (const float*)d_in[6];
  const float* ln1_w = (const float*)d_in[7];
  const float* ln1_b = (const float*)d_in[8];
  const float* ff_W1 = (const float*)d_in[9];
  const float* ff_b1 = (const float*)d_in[10];
  const float* ff_W2 = (const float*)d_in[11];
  const float* ff_b2 = (const float*)d_in[12];
  const float* ln2_w = (const float*)d_in[13];
  const float* ln2_b = (const float*)d_in[14];
  float* out = (float*)d_out;

  char* ws = (char*)d_ws;
  size_t off = 0;
  auto alloc = [&](size_t bytes) -> char* {
    char* p = ws + off;
    off = (off + bytes + 255) & ~(size_t)255;
    return p;
  };
  unsigned short* xb = (unsigned short*)alloc((size_t)BS * D_MODEL * 2);
  unsigned short* wqkv = (unsigned short*)alloc((size_t)NUM_EXPERTS * D_MODEL * QKV3 * 2);
  unsigned short* wout = (unsigned short*)alloc((size_t)D_MODEL * D_MODEL * 2);
  unsigned short* wff1 = (unsigned short*)alloc((size_t)D_MODEL * D_FF * 2);
  unsigned short* wff2 = (unsigned short*)alloc((size_t)D_FF * D_MODEL * 2);
  float* routing = (float*)alloc((size_t)BS * NUM_EXPERTS * 4);
  unsigned short* qb = (unsigned short*)alloc((size_t)BATCH * NUM_EXPERTS * SEQ * HEAD_DIM * 2);
  unsigned short* kb = (unsigned short*)alloc((size_t)BATCH * NUM_EXPERTS * SEQ * HEAD_DIM * 2);
  unsigned short* vb = (unsigned short*)alloc((size_t)BATCH * NUM_EXPERTS * SEQ * HEAD_DIM * 2);
  unsigned short* comb = (unsigned short*)alloc((size_t)BS * D_MODEL * 2);
  float* pre = (float*)alloc((size_t)BS * D_MODEL * 4);
  float* x1f = (float*)alloc((size_t)BS * D_MODEL * 4);
  unsigned short* x1b = (unsigned short*)alloc((size_t)BS * D_MODEL * 2);
  unsigned short* hb = (unsigned short*)alloc((size_t)BS * D_FF * 2);

  cast_kernel<<<2048, 256, 0, stream>>>(x, xb, (long)BS * D_MODEL);
  cast_kernel<<<2048, 256, 0, stream>>>(expert_W, wqkv, (long)NUM_EXPERTS * D_MODEL * QKV3);
  cast_kernel<<<2048, 256, 0, stream>>>(out_W, wout, (long)D_MODEL * D_MODEL);
  cast_kernel<<<2048, 256, 0, stream>>>(ff_W1, wff1, (long)D_MODEL * D_FF);
  cast_kernel<<<2048, 256, 0, stream>>>(ff_W2, wff2, (long)D_FF * D_MODEL);

  router_kernel<<<BS, 32, 0, stream>>>(x, router_W, router_b, routing);

  // all-expert QKV: x[4096,1280] @ W_e[1280,192] -> q/k/v [B,E,S,64] bf16
  gemm_kernel<1, 0, 0, 0, 1><<<dim3(QKV3 / 64, BS / 128, NUM_EXPERTS), 256, 0, stream>>>(
      xb, wqkv, expert_b, nullptr, nullptr, nullptr, qb, kb, vb, BS, QKV3, D_MODEL);

  attn_kernel<<<dim3(SEQ / 64, NUM_EXPERTS, BATCH), 128, 0, stream>>>(qb, kb, vb, routing, comb);

  // out projection + residual(x) -> pre (f32)
  gemm_kernel<0, 0, 1, 1, 0><<<dim3(D_MODEL / 64, BS / 128), 256, 0, stream>>>(
      comb, wout, out_b, x, pre, nullptr, nullptr, nullptr, nullptr, BS, D_MODEL, D_MODEL);

  ln_kernel<<<BS, 128, 0, stream>>>(pre, ln1_w, ln1_b, x1f, x1b);

  // FF1 + GELU -> hb (bf16)
  gemm_kernel<0, 1, 0, 0, 1><<<dim3(D_FF / 64, BS / 128), 256, 0, stream>>>(
      x1b, wff1, ff_b1, nullptr, nullptr, hb, nullptr, nullptr, nullptr, BS, D_FF, D_MODEL);

  // FF2 + residual(x1) -> pre (f32)
  gemm_kernel<0, 0, 1, 1, 0><<<dim3(D_MODEL / 64, BS / 128), 256, 0, stream>>>(
      hb, wff2, ff_b2, x1f, pre, nullptr, nullptr, nullptr, nullptr, BS, D_MODEL, D_FF);

  ln_kernel<<<BS, 128, 0, stream>>>(pre, ln2_w, ln2_b, out, nullptr);
}
